// GNNModel_53120155517251
// MI455X (gfx1250) — matrix operand order repaired and checked by reading, not compiled, not run on hardware
//
#include <hip/hip_runtime.h>

#define N_ATOMS 32000
#define GRAPHS  32
#define HID     128
#define NG      50
#define LAYERS  6

typedef __attribute__((ext_vector_type(16))) __bf16        bf16x16;
typedef __attribute__((ext_vector_type(8)))  float         floatx8;
typedef __attribute__((ext_vector_type(8)))  unsigned int  uintx8;

// ---------- helpers ----------

// shifted softplus: log(1+e^x) - log(2), numerically stable
static __device__ inline float ssp(float v) {
    float t = fabsf(v);
    float r = __logf(1.0f + __expf(-t));
    return (v > 0.0f ? v + r : r) - 0.69314718056f;
}

// raw 32-bit LDS byte offset of a generic pointer that points into __shared__
static __device__ inline unsigned lds_offset(const void* p) {
    return (unsigned)(unsigned long long)
        (__attribute__((address_space(3))) const unsigned char*)p;
}

// A fragment (16x32 bf16, MxK) from an LDS f32 tile [16][ldk], row m = lane&15.
// ISA layout: element i covers K = kbase + (i/8)*16 + (lane/16)*8 + (i%8)
// -> two runs of 8 consecutive floats (ds_load_b128-mergeable) + hw bf16 cvt.
static __device__ inline bf16x16 load_a_lds(const float* lds, int lane, int kbase, int ldk) {
    int m  = lane & 15;
    int kh = (lane >> 4) << 3;
    const float* p0 = lds + m * ldk + kbase + kh;
    bf16x16 a;
#pragma unroll
    for (int i = 0; i < 8; ++i) a[i]     = (__bf16)p0[i];
#pragma unroll
    for (int i = 0; i < 8; ++i) a[i + 8] = (__bf16)p0[16 + i];
    return a;
}

// B fragment from pre-packed bf16 weights (fragment-major): 32 bytes per lane,
// two coalesced global b128 loads, no conversion, no predication.
static __device__ inline bf16x16 load_b_packed(const __bf16* __restrict__ p, int lane, int frag) {
    uintx8 u = *(const uintx8*)(p + (((size_t)frag << 5) + lane) * 16);
    return __builtin_bit_cast(bf16x16, u);
}

// 16x128 f32 tile: global rows [row0, row0+16) -> LDS via CDNA5 async DMA
// (global_load_async_to_lds_b128, tracked by ASYNCcnt; no VGPR round-trip).
// LDS offset and global address both advance by t*512B -> shared INST_OFFSET.
static __device__ inline void stage_tile(const float* __restrict__ g, float* lds,
                                         int row0, int lane) {
    unsigned loff = lds_offset(lds) + (unsigned)lane * 16u;
    unsigned voff = (unsigned)lane * 16u;
    unsigned long long base = (unsigned long long)(g + (size_t)row0 * HID);
#pragma unroll
    for (int t = 0; t < 16; ++t) {
        asm volatile("global_load_async_to_lds_b128 %0, %1, %2 offset:%3"
                     :: "v"(loff), "v"(voff), "s"(base), "n"(t * 512)
                     : "memory");
    }
    asm volatile("s_wait_asynccnt 0x0" ::: "memory");
}

#define WMMA_BF16(A, B, C) \
    __builtin_amdgcn_wmma_f32_16x16x32_bf16(false, (A), false, (B), (short)0, (C), false, false)

// ---------- weight pre-pack ----------
// src: [K x N] f32 row-major.  dst: fragment-major bf16, zero-padded to Kpad.
// dst[((nt*(Kpad/32)+kb)*32 + lane)*16 + i] = bf16(src[k*N+n]),
//   k = kb*32 + (i/8)*16 + (lane/16)*8 + (i%8),  n = nt*16 + (lane&15)   (ISA B layout)
__global__ void k_pack(const float* __restrict__ src, __bf16* __restrict__ dst,
                       int K, int Kpad, int N) {
    int idx   = blockIdx.x * blockDim.x + threadIdx.x;
    int total = (N >> 4) * (Kpad >> 5) * 512;
    if (idx >= total) return;
    int i    = idx & 15;
    int lane = (idx >> 4) & 31;
    int frag = idx >> 9;
    int KB   = Kpad >> 5;
    int kb   = frag % KB;
    int nt   = frag / KB;
    int k = kb * 32 + ((i >> 3) << 4) + ((lane >> 4) << 3) + (i & 7);
    int n = nt * 16 + (lane & 15);
    float v = (k < K) ? src[k * N + n] : 0.0f;
    dst[idx] = (__bf16)v;
}

// ---------- kernels ----------

__global__ void k_embed(const int* __restrict__ z, const float* __restrict__ emb,
                        float* __restrict__ h, int n) {
    int i = blockIdx.x * blockDim.x + threadIdx.x;
    if (i < n * HID) {
        int a = i >> 7, f = i & 127;
        h[i] = emb[z[a] * HID + f];
    }
}

// x = h @ lin1_w   (per-wave 16x128 tile)
__global__ __launch_bounds__(32) void k_lin1(const float* __restrict__ h,
                                             const __bf16* __restrict__ pw,
                                             float* __restrict__ x) {
    __shared__ float tile[16 * HID];
    int lane = threadIdx.x;
    int row0 = blockIdx.x * 16;
    stage_tile(h, tile, row0, lane);
    __syncthreads();
    bf16x16 afr[4];
#pragma unroll
    for (int kb = 0; kb < 4; ++kb) afr[kb] = load_a_lds(tile, lane, kb * 32, HID);
    int half = lane >> 4, nl = lane & 15;
    for (int nt = 0; nt < 8; ++nt) {
        floatx8 acc = {};
#pragma unroll
        for (int kb = 0; kb < 4; ++kb)
            acc = WMMA_BF16(afr[kb], load_b_packed(pw, lane, nt * 4 + kb), acc);
#pragma unroll
        for (int r = 0; r < 8; ++r)
            x[(row0 + r + 8 * half) * HID + nt * 16 + nl] = acc[r];
    }
}

// fused: rbf -> filter MLP (two WMMA GEMMs) -> cutoff -> gather x[col] -> scatter-add agg[row]
__global__ __launch_bounds__(32) void k_edge(
    const float* __restrict__ pos, const int* __restrict__ row, const int* __restrict__ col,
    const __bf16* __restrict__ pw1, const float* __restrict__ b1,
    const __bf16* __restrict__ pw2, const float* __restrict__ b2,
    const float* __restrict__ x, float* __restrict__ agg, int E) {
    __shared__ float rbf[16 * 64];     // NG=50; cols 50..63 never matter (B rows zero-padded)
    __shared__ float tmid[16 * HID];
    __shared__ float sd[16], sC[16];
    __shared__ int srow[16], scol[16];
    int lane = threadIdx.x;
    int e0 = blockIdx.x * 16;

    if (lane < 16) {
        int e = e0 + lane;
        bool valid = e < E;
        int ec = valid ? e : (E - 1);
        int r = row[ec], c = col[ec];
        float dx = pos[r * 3 + 0] - pos[c * 3 + 0];
        float dy = pos[r * 3 + 1] - pos[c * 3 + 1];
        float dz = pos[r * 3 + 2] - pos[c * 3 + 2];
        float d = sqrtf(dx * dx + dy * dy + dz * dz);
        sd[lane] = d;
        sC[lane] = valid ? 0.5f * (__cosf(d * 0.62831853072f) + 1.0f) : 0.0f;  // pi/CUTOFF
        srow[lane] = r; scol[lane] = c;
    }
    __syncthreads();

    // prefetch the 16 gather rows (4 x 128B lines each) ~2 GEMMs ahead of use
    {
        const float* pf = x + (size_t)scol[lane & 15] * HID + (lane >> 4) * 64;
        __builtin_prefetch(pf, 0, 1);        // bytes 0 / 256 of the row
        __builtin_prefetch(pf + 32, 0, 1);   // bytes 128 / 384
    }

    const float step  = 5.0f / 49.0f;
    const float coeff = -0.5f / (step * step);
#pragma unroll
    for (int t = 0; t < 32; ++t) {
        int idx = lane + 32 * t;            // 16*64 entries
        int e = idx >> 6, g = idx & 63;
        float diff = sd[e] - (float)g * step;
        rbf[idx] = (g < NG) ? __expf(coeff * diff * diff) : 0.0f;
    }
    __syncthreads();

    int half = lane >> 4, nl = lane & 15;

    // GEMM1: [16x64(rbf)] @ w1[64(pad)x128] + b1 -> ssp -> tmid
    bf16x16 afr[2];
#pragma unroll
    for (int kb = 0; kb < 2; ++kb) afr[kb] = load_a_lds(rbf, lane, kb * 32, 64);
    for (int nt = 0; nt < 8; ++nt) {
        floatx8 acc = {};
#pragma unroll
        for (int kb = 0; kb < 2; ++kb)
            acc = WMMA_BF16(afr[kb], load_b_packed(pw1, lane, nt * 2 + kb), acc);
        float bias = b1[nt * 16 + nl];
#pragma unroll
        for (int r = 0; r < 8; ++r)
            tmid[(r + 8 * half) * HID + nt * 16 + nl] = ssp(acc[r] + bias);
    }
    __syncthreads();

    // GEMM2: tmid @ w2[128x128] + b2, * C, gather x[col], scatter-add to agg[row]
    bf16x16 tfr[4];
#pragma unroll
    for (int kb = 0; kb < 4; ++kb) tfr[kb] = load_a_lds(tmid, lane, kb * 32, HID);
    for (int nt = 0; nt < 8; ++nt) {
        floatx8 acc = {};
#pragma unroll
        for (int kb = 0; kb < 4; ++kb)
            acc = WMMA_BF16(tfr[kb], load_b_packed(pw2, lane, nt * 4 + kb), acc);
        int n = nt * 16 + nl;
        float bias = b2[n];
#pragma unroll
        for (int r = 0; r < 8; ++r) {
            int m = r + 8 * half;
            float Wv  = (acc[r] + bias) * sC[m];        // sC==0 kills padded edges
            float val = x[scol[m] * HID + n] * Wv;
            atomicAdd(&agg[srow[m] * HID + n], val);
        }
    }
}

// v = ssp(agg @ lin2_w + lin2_b); h += v @ lin_w + lin_b
__global__ __launch_bounds__(32) void k_update(const float* __restrict__ agg,
    const __bf16* __restrict__ pw2, const float* __restrict__ b2,
    const __bf16* __restrict__ pw3, const float* __restrict__ b3,
    float* __restrict__ h) {
    __shared__ float tile[16 * HID];
    __shared__ float vtile[16 * HID];
    int lane = threadIdx.x;
    int row0 = blockIdx.x * 16;
    stage_tile(agg, tile, row0, lane);
    __syncthreads();
    int half = lane >> 4, nl = lane & 15;
    bf16x16 afr[4];
#pragma unroll
    for (int kb = 0; kb < 4; ++kb) afr[kb] = load_a_lds(tile, lane, kb * 32, HID);
    for (int nt = 0; nt < 8; ++nt) {
        floatx8 acc = {};
#pragma unroll
        for (int kb = 0; kb < 4; ++kb)
            acc = WMMA_BF16(afr[kb], load_b_packed(pw2, lane, nt * 4 + kb), acc);
        float bias = b2[nt * 16 + nl];
#pragma unroll
        for (int r = 0; r < 8; ++r)
            vtile[(r + 8 * half) * HID + nt * 16 + nl] = ssp(acc[r] + bias);
    }
    __syncthreads();
    bf16x16 vfr[4];
#pragma unroll
    for (int kb = 0; kb < 4; ++kb) vfr[kb] = load_a_lds(vtile, lane, kb * 32, HID);
    for (int nt = 0; nt < 8; ++nt) {
        floatx8 acc = {};
#pragma unroll
        for (int kb = 0; kb < 4; ++kb)
            acc = WMMA_BF16(vfr[kb], load_b_packed(pw3, lane, nt * 4 + kb), acc);
        float bias = b3[nt * 16 + nl];
#pragma unroll
        for (int r = 0; r < 8; ++r) {
            int idx = (row0 + r + 8 * half) * HID + nt * 16 + nl;
            h[idx] += acc[r] + bias;
        }
    }
}

// a = ssp(h@out1_w+b); per_atom = a@out2_w+b; atomic segment-sum into sch[batch]
__global__ __launch_bounds__(32) void k_out(const float* __restrict__ h,
    const __bf16* __restrict__ pw1, const float* __restrict__ b1,
    const float* __restrict__ w2, const float* __restrict__ b2,
    const int* __restrict__ batch, float* __restrict__ sch) {
    __shared__ float tile[16 * HID];
    __shared__ float atile[16 * 64];
    int lane = threadIdx.x;
    int row0 = blockIdx.x * 16;
    stage_tile(h, tile, row0, lane);
    __syncthreads();
    int half = lane >> 4, nl = lane & 15;
    bf16x16 afr[4];
#pragma unroll
    for (int kb = 0; kb < 4; ++kb) afr[kb] = load_a_lds(tile, lane, kb * 32, HID);
    for (int nt = 0; nt < 4; ++nt) {
        floatx8 acc = {};
#pragma unroll
        for (int kb = 0; kb < 4; ++kb)
            acc = WMMA_BF16(afr[kb], load_b_packed(pw1, lane, nt * 4 + kb), acc);
        float bias = b1[nt * 16 + nl];
#pragma unroll
        for (int r = 0; r < 8; ++r)
            atile[(r + 8 * half) * 64 + nt * 16 + nl] = ssp(acc[r] + bias);
    }
    __syncthreads();
    if (lane < 16) {
        float s = b2[0];
#pragma unroll 8
        for (int k = 0; k < 64; ++k) s += atile[lane * 64 + k] * w2[k];
        atomicAdd(&sch[batch[row0 + lane]], s);
    }
}

// per-graph head: relu(sch*head1_w+b1) @ head2_w + b2
__global__ __launch_bounds__(32) void k_head(const float* __restrict__ sch,
    const float* __restrict__ h1w, const float* __restrict__ h1b,
    const float* __restrict__ h2w, const float* __restrict__ h2b,
    float* __restrict__ out) {
    int g = threadIdx.x;
    if (g < GRAPHS) {
        float s = sch[g];
        float acc = h2b[0];
#pragma unroll 8
        for (int j = 0; j < 64; ++j) {
            float v = s * h1w[j] + h1b[j];
            acc += (v > 0.0f ? v : 0.0f) * h2w[j];
        }
        out[g] = acc;
    }
}

// ---------- launcher ----------

#define PW1_ELEMS  (8 * 2 * 512)    // 64x128  -> 8192 bf16
#define PW_ELEMS   (8 * 4 * 512)    // 128x128 -> 16384 bf16
#define POUT_ELEMS (4 * 4 * 512)    // 128x64  -> 8192 bf16

extern "C" void kernel_launch(void* const* d_in, const int* in_sizes, int n_in,
                              void* d_out, int out_size, void* d_ws, size_t ws_size,
                              hipStream_t stream) {
    const int*   z       = (const int*)  d_in[0];
    const float* pos     = (const float*)d_in[1];
    const int*   batch   = (const int*)  d_in[2];
    const int*   row     = (const int*)  d_in[3];
    const int*   col     = (const int*)  d_in[4];
    const float* emb     = (const float*)d_in[5];
    const float* mlp_w1  = (const float*)d_in[6];
    const float* mlp_b1  = (const float*)d_in[7];
    const float* mlp_w2  = (const float*)d_in[8];
    const float* mlp_b2  = (const float*)d_in[9];
    const float* lin1_w  = (const float*)d_in[10];
    const float* lin2_w  = (const float*)d_in[11];
    const float* lin2_b  = (const float*)d_in[12];
    const float* lin_w   = (const float*)d_in[13];
    const float* lin_b   = (const float*)d_in[14];
    const float* out1_w  = (const float*)d_in[15];
    const float* out1_b  = (const float*)d_in[16];
    const float* out2_w  = (const float*)d_in[17];
    const float* out2_b  = (const float*)d_in[18];
    const float* head1_w = (const float*)d_in[19];
    const float* head1_b = (const float*)d_in[20];
    const float* head2_w = (const float*)d_in[21];
    const float* head2_b = (const float*)d_in[22];

    const int E = in_sizes[3];

    float* h   = (float*)d_ws;
    float* x   = h   + (size_t)N_ATOMS * HID;
    float* agg = x   + (size_t)N_ATOMS * HID;
    float* sch = agg + (size_t)N_ATOMS * HID;
    __bf16* pk = (__bf16*)(sch + 64);          // packed-weight region (bf16)
    float* out = (float*)d_out;

    // per-layer packed slices: [pw1 | plin1 | plin2 | plin | pw2] x LAYERS, then pout1
    const size_t LSTRIDE = PW1_ELEMS + 4 * (size_t)PW_ELEMS;
    __bf16* pout1 = pk + LAYERS * LSTRIDE;

    // ---- pack all weights to bf16, fragment-major (L2-resident, ~0.9 MB) ----
    for (int l = 0; l < LAYERS; ++l) {
        __bf16* base = pk + l * LSTRIDE;
        k_pack<<<(PW1_ELEMS + 255) / 256, 256, 0, stream>>>(
            mlp_w1 + (size_t)l * NG * HID, base, NG, 64, HID);
        k_pack<<<(PW_ELEMS + 255) / 256, 256, 0, stream>>>(
            lin1_w + (size_t)l * HID * HID, base + PW1_ELEMS, HID, HID, HID);
        k_pack<<<(PW_ELEMS + 255) / 256, 256, 0, stream>>>(
            lin2_w + (size_t)l * HID * HID, base + PW1_ELEMS + PW_ELEMS, HID, HID, HID);
        k_pack<<<(PW_ELEMS + 255) / 256, 256, 0, stream>>>(
            lin_w + (size_t)l * HID * HID, base + PW1_ELEMS + 2 * PW_ELEMS, HID, HID, HID);
        k_pack<<<(PW_ELEMS + 255) / 256, 256, 0, stream>>>(
            mlp_w2 + (size_t)l * HID * HID, base + PW1_ELEMS + 3 * PW_ELEMS, HID, HID, HID);
    }
    k_pack<<<(POUT_ELEMS + 255) / 256, 256, 0, stream>>>(out1_w, pout1, HID, HID, 64);

    // ---- network ----
    k_embed<<<(N_ATOMS * HID + 255) / 256, 256, 0, stream>>>(z, emb, h, N_ATOMS);

    for (int l = 0; l < LAYERS; ++l) {
        __bf16* base  = pk + l * LSTRIDE;
        __bf16* pw1   = base;
        __bf16* plin1 = base + PW1_ELEMS;
        __bf16* plin2 = base + PW1_ELEMS + PW_ELEMS;
        __bf16* plin  = base + PW1_ELEMS + 2 * PW_ELEMS;
        __bf16* pw2   = base + PW1_ELEMS + 3 * PW_ELEMS;

        hipMemsetAsync(agg, 0, (size_t)N_ATOMS * HID * sizeof(float), stream);
        k_lin1<<<N_ATOMS / 16, 32, 0, stream>>>(h, plin1, x);
        k_edge<<<(E + 15) / 16, 32, 0, stream>>>(
            pos, row, col,
            pw1, mlp_b1 + (size_t)l * HID,
            pw2, mlp_b2 + (size_t)l * HID,
            x, agg, E);
        k_update<<<N_ATOMS / 16, 32, 0, stream>>>(
            agg,
            plin2, lin2_b + (size_t)l * HID,
            plin,  lin_b  + (size_t)l * HID,
            h);
    }

    hipMemsetAsync(sch, 0, GRAPHS * sizeof(float), stream);
    k_out<<<N_ATOMS / 16, 32, 0, stream>>>(h, pout1, out1_b, out2_w, out2_b, batch, sch);
    k_head<<<1, 32, 0, stream>>>(sch, head1_w, head1_b, head2_w, head2_b, out);
}
